// Attention_11793980194868
// MI455X (gfx1250) — matrix operand order repaired and checked by reading, not compiled, not run on hardware
//
#include <hip/hip_runtime.h>
#include <hip/hip_bf16.h>

typedef __bf16 bf16;
typedef __attribute__((ext_vector_type(16))) __bf16 v16bf;
typedef __attribute__((ext_vector_type(8)))  __bf16 v8bf;
typedef __attribute__((ext_vector_type(8)))  float  v8f;
typedef int v4i __attribute__((vector_size(16)));   // b128 payload type
typedef __attribute__((address_space(1))) v4i GV4;  // global
typedef __attribute__((address_space(3))) v4i LV4;  // LDS

#define DEVINL static __device__ __forceinline__

// ---------------------------------------------------------------------------
// CDNA5 async global->LDS staging (ASYNCcnt path, CDNA5 ISA)
// Guarded: falls back to sync load + ds_store if the builtin is absent.
// ---------------------------------------------------------------------------
#if defined(__has_builtin)
#if __has_builtin(__builtin_amdgcn_global_load_async_to_lds_b128)
#define ASYNC_LDS 1
#endif
#endif

DEVINL void stage16(const bf16* g, bf16* l) {
#ifdef ASYNC_LDS
    __builtin_amdgcn_global_load_async_to_lds_b128((GV4*)g, (LV4*)l, 0, 0);
#else
    *(v8bf*)l = *(const v8bf*)g;
#endif
}

DEVINL void async_join() {
#ifdef ASYNC_LDS
    asm volatile("s_wait_asynccnt 0" ::: "memory");
#endif
}

// ---------------------------------------------------------------------------
// WMMA fragment helpers (CDNA5 ISA layouts)
// A (16x32, MxK): lane = m + 16*half, two 8-elem chunks at K=8*half, 16+8*half
// B (32x16, KxN): lane = n + 16*khalf, same K order as A (GEMM: 16 contiguous K at 16*khalf)
// C (16x16 f32):  lane = n + 16*hi, VGPR r -> row r + 8*hi
// ---------------------------------------------------------------------------
DEVINL v16bf load_frag2(const bf16* p0, const bf16* p1) {
    v8bf a = *(const v8bf*)p0;
    v8bf b = *(const v8bf*)p1;
    v16bf r;
#pragma unroll
    for (int i = 0; i < 8; ++i) { r[i] = a[i]; r[i + 8] = b[i]; }
    return r;
}

DEVINL v8f wmma_bf16(v16bf a, v16bf b, v8f c) {
    return __builtin_amdgcn_wmma_f32_16x16x32_bf16(false, a, false, b,
                                                   (short)0, c, false, false);
}

// ---------------------------------------------------------------------------
// Kernel 1: depthwise conv3d (k=3, pad=1, groups=C) + BatchNorm (eval) -> bf16
// ---------------------------------------------------------------------------
__global__ void dwconv_bn_kernel(const float* __restrict__ x,
                                 const float* __restrict__ w,
                                 const float* __restrict__ g,
                                 const float* __restrict__ beta,
                                 const float* __restrict__ mean,
                                 const float* __restrict__ var,
                                 bf16* __restrict__ out,
                                 int stride, int S) {
    const int C = 384;
    long long idx = (long long)blockIdx.x * blockDim.x + threadIdx.x;
    long long total = 8LL * S * S * S * C;
    if (idx >= total) return;

    int c  = (int)(idx % C);
    long long t = idx / C;
    int S3 = S * S * S;
    int sp = (int)(t % S3);
    int b  = (int)(t / S3);
    int od = sp % S;
    int ow = (sp / S) % S;
    int oh = sp / (S * S);

    const float* wc = w + c * 27;
    float acc = 0.f;
#pragma unroll
    for (int kh = 0; kh < 3; ++kh) {
        int ih = oh * stride + kh - 1;
        if ((unsigned)ih >= 16u) continue;
#pragma unroll
        for (int kw = 0; kw < 3; ++kw) {
            int iw = ow * stride + kw - 1;
            if ((unsigned)iw >= 16u) continue;
#pragma unroll
            for (int kd = 0; kd < 3; ++kd) {
                int id = od * stride + kd - 1;
                if ((unsigned)id >= 16u) continue;
                long long xi = ((long long)b * 4096 +
                                ((ih * 16 + iw) * 16 + id)) * C + c;
                acc += x[xi] * wc[(kh * 3 + kw) * 3 + kd];
            }
        }
    }
    float inv = g[c] * rsqrtf(var[c] + 1e-5f);
    float y = acc * inv + (beta[c] - mean[c] * inv);
    out[idx] = (bf16)y;
}

// ---------------------------------------------------------------------------
// Kernel 2: f32 -> bf16 conversion (weights)
// ---------------------------------------------------------------------------
__global__ void f32_to_bf16_kernel(const float* __restrict__ in,
                                   bf16* __restrict__ out, int n) {
    int i = blockIdx.x * blockDim.x + threadIdx.x;
    if (i < n) out[i] = (bf16)in[i];
}

// ---------------------------------------------------------------------------
// Kernel 3: NT GEMM  Y[M,N] = A[M,K] * W[N,K]^T, K = N = 384.
// 256 threads / 8 waves; block tile 128(M) x 64(N); the 64x384 weight tile is
// async-staged to LDS once and shared by all 8 waves (8x VMEM cut).
// MODE 0: bf16 out [M,N]; MODE 1: bf16 out as V^T [b,h,dh,512]; MODE 2: f32+bias
// ---------------------------------------------------------------------------
template <int MODE>
__global__ void gemm_nt_wmma(const bf16* __restrict__ A,
                             const bf16* __restrict__ Bw,
                             void* __restrict__ outp,
                             const float* __restrict__ bias,
                             int M, int N, int K) {
    __shared__ bf16 sB[64 * 384];   // 48 KB
    const int tid  = threadIdx.x;
    const int lane = tid & 31;
    const int wave = tid >> 5;
    const int n15  = lane & 15;
    const int hi   = lane >> 4;
    const int m0   = blockIdx.y * 128 + wave * 16;
    const int n0   = blockIdx.x * 64;

    // cooperative stage of W[n0..n0+63, 0..383] -> LDS (3072 x 16B)
    for (int e = tid; e < 64 * 384 / 8; e += 256) {
        int row = e / 48;
        int off = (e % 48) * 8;
        stage16(Bw + (size_t)(n0 + row) * K + off, sB + row * 384 + off);
    }
    async_join();
    __syncthreads();

    v8f acc[4];
#pragma unroll
    for (int j = 0; j < 4; ++j) { v8f z = {}; acc[j] = z; }

    const bf16* arow = A + (size_t)(m0 + n15) * K;
    for (int kk = 0; kk < K; kk += 32) {
        const bf16* ap = arow + kk + 16 * hi;
        v16bf af = load_frag2(ap, ap + 8);
        const bf16* bp = sB + n15 * 384 + kk + 16 * hi;
#pragma unroll
        for (int j = 0; j < 4; ++j) {
            const bf16* bj = bp + j * 16 * 384;
            acc[j] = wmma_bf16(af, load_frag2(bj, bj + 8), acc[j]);
        }
    }

#pragma unroll
    for (int j = 0; j < 4; ++j) {
        int col = n0 + j * 16 + n15;
#pragma unroll
        for (int r = 0; r < 8; ++r) {
            int row = m0 + r + 8 * hi;
            float val = acc[j][r];
            if (MODE == 0) {
                ((bf16*)outp)[(size_t)row * N + col] = (bf16)val;
            } else if (MODE == 1) {
                int b = row >> 9, t = row & 511;
                int head = col >> 6, dh = col & 63;
                ((bf16*)outp)[(size_t)((b * 6 + head) * 64 + dh) * 512 + t] =
                    (bf16)val;
            } else {
                ((float*)outp)[(size_t)row * N + col] = val + bias[col];
            }
        }
    }
}

// ---------------------------------------------------------------------------
// Kernel 4: flash attention. Block = 4 waves = 4 query tiles of one (b,head).
// K/V^T are staged in double-buffered 128-kv LDS chunks (async prefetch of
// chunk c+1 overlaps compute of chunk c); shared by the 4 waves (4x VMEM cut).
// S^T = K*Q^T orientation keeps softmax stats per-lane (+1 shfl_xor(16)).
// ---------------------------------------------------------------------------
__global__ void attn_kernel(const bf16* __restrict__ Q,
                            const bf16* __restrict__ Kmat,
                            const bf16* __restrict__ Vt,
                            bf16* __restrict__ O) {
    const int Cd = 384, T = 4096, TKV = 512, H = 6, CH = 128;
    const float scale = 0.05103103630798288f;  // 384^-0.5

    extern __shared__ bf16 smem[];
    bf16* sK = smem;                // [2][CH][64]
    bf16* sV = smem + 2 * CH * 64;  // [2][64][CH]

    int qblk = blockIdx.x & 63;
    int bh   = blockIdx.x >> 6;
    int head = bh % H, b = bh / H;
    int tid  = threadIdx.x;
    int wave = tid >> 5;
    int lane = tid & 31;
    int qt   = qblk * 4 + wave;
    int n = lane & 15, hi = lane >> 4;

    const bf16* kb  = Kmat + (size_t)b * TKV * Cd + head * 64;
    const bf16* vtb = Vt + ((size_t)(b * H + head) * 64) * TKV;

    auto stage_chunk = [&](int buf, int kv0) {
        bf16* dK = sK + buf * CH * 64;
        bf16* dV = sV + buf * 64 * CH;
        for (int e = tid; e < CH * 64 / 8; e += 128) {      // K: CH rows x 64
            int row = e >> 3, off = (e & 7) * 8;
            stage16(kb + (size_t)(kv0 + row) * Cd + off, dK + row * 64 + off);
        }
        for (int e = tid; e < 64 * CH / 8; e += 128) {      // V^T: 64 rows x CH
            int row = e >> 4, off = (e & 15) * 8;
            stage16(vtb + (size_t)row * TKV + kv0 + off, dV + row * CH + off);
        }
    };

    // resident Q^T B-fragments (dh chunks at 8*hi, 16+8*hi per lane)
    const bf16* qp = Q + ((size_t)b * T + qt * 16 + n) * Cd + head * 64 + 8 * hi;
    v16bf qb0 = load_frag2(qp, qp + 16);       // dh 0..31
    v16bf qb1 = load_frag2(qp + 32, qp + 48);  // dh 32..63

    v8f o0 = {}, o1 = {}, o2 = {}, o3 = {};
    float mprev = -1e30f, lsum = 0.f;

    stage_chunk(0, 0);
    async_join();
    __syncthreads();

    for (int c = 0; c < TKV / CH; ++c) {
        if (c + 1 < TKV / CH) stage_chunk((c + 1) & 1, (c + 1) * CH);
        const bf16* bK = sK + (c & 1) * CH * 64;
        const bf16* bV = sV + (c & 1) * 64 * CH;

        for (int kv = 0; kv < CH; kv += 32) {
            // ---- S^T tiles (16kv x 16q f32), K fragments from LDS ----
            v8f s0 = {}, s1 = {};
            const bf16* kr = bK + (kv + n) * 64 + 8 * hi;
            s0 = wmma_bf16(load_frag2(kr, kr + 16), qb0, s0);
            s0 = wmma_bf16(load_frag2(kr + 32, kr + 48), qb1, s0);
            const bf16* kr2 = kr + 16 * 64;
            s1 = wmma_bf16(load_frag2(kr2, kr2 + 16), qb0, s1);
            s1 = wmma_bf16(load_frag2(kr2 + 32, kr2 + 48), qb1, s1);

            // ---- online softmax over this lane's query column ----
            float mloc = -1e30f;
#pragma unroll
            for (int r = 0; r < 8; ++r) {
                s0[r] *= scale; s1[r] *= scale;
                mloc = fmaxf(mloc, fmaxf(s0[r], s1[r]));
            }
            mloc = fmaxf(mloc, __shfl_xor(mloc, 16, 32));
            float mnew = fmaxf(mprev, mloc);
            float corr = __expf(mprev - mnew);
            float p0[8], p1[8], psum = 0.f;
#pragma unroll
            for (int r = 0; r < 8; ++r) {
                p0[r] = __expf(s0[r] - mnew);
                p1[r] = __expf(s1[r] - mnew);
                psum += p0[r] + p1[r];
            }
            lsum = lsum * corr + psum;
#pragma unroll
            for (int r = 0; r < 8; ++r) {
                o0[r] *= corr; o1[r] *= corr; o2[r] *= corr; o3[r] *= corr;
            }
            mprev = mnew;

            // ---- P^T B-fragment (32kv x 16q) from two C-layout tiles ----
            v16bf pb;
#pragma unroll
            for (int r = 0; r < 8; ++r) {
                float a0 = __shfl_xor(p0[r], 16, 32);
                float a1 = __shfl_xor(p1[r], 16, 32);
                pb[r]     = (bf16)p0[r];
                pb[r + 8] = (bf16)p1[r];
            }

            // ---- O^T += V^T * P^T (4 dh tiles), V fragments from LDS ----
            const bf16* vr = bV + n * CH + kv + 8 * hi;
            o0 = wmma_bf16(load_frag2(vr, vr + 16), pb, o0);
            o1 = wmma_bf16(load_frag2(vr + 16 * CH, vr + 16 * CH + 16), pb, o1);
            o2 = wmma_bf16(load_frag2(vr + 32 * CH, vr + 32 * CH + 16), pb, o2);
            o3 = wmma_bf16(load_frag2(vr + 48 * CH, vr + 48 * CH + 16), pb, o3);
        }
        async_join();
        __syncthreads();
    }

    lsum += __shfl_xor(lsum, 16, 32);
    float rinv = 1.f / lsum;
    bf16* op = O + ((size_t)b * T + qt * 16 + n) * Cd + head * 64;
#pragma unroll
    for (int r = 0; r < 8; ++r) {
        op[ 0 + r + 8 * hi] = (bf16)(o0[r] * rinv);
        op[16 + r + 8 * hi] = (bf16)(o1[r] * rinv);
        op[32 + r + 8 * hi] = (bf16)(o2[r] * rinv);
        op[48 + r + 8 * hi] = (bf16)(o3[r] * rinv);
    }
}

// ---------------------------------------------------------------------------
extern "C" void kernel_launch(void* const* d_in, const int* in_sizes, int n_in,
                              void* d_out, int out_size, void* d_ws, size_t ws_size,
                              hipStream_t stream) {
    const float* x    = (const float*)d_in[0];
    const float* cqw  = (const float*)d_in[1];
    const float* ckw  = (const float*)d_in[2];
    const float* cvw  = (const float*)d_in[3];
    const float* bq_g = (const float*)d_in[4];
    const float* bq_b = (const float*)d_in[5];
    const float* bq_m = (const float*)d_in[6];
    const float* bq_v = (const float*)d_in[7];
    const float* bk_g = (const float*)d_in[8];
    const float* bk_b = (const float*)d_in[9];
    const float* bk_m = (const float*)d_in[10];
    const float* bk_v = (const float*)d_in[11];
    const float* bv_g = (const float*)d_in[12];
    const float* bv_b = (const float*)d_in[13];
    const float* bv_m = (const float*)d_in[14];
    const float* bv_v = (const float*)d_in[15];
    const float* wq   = (const float*)d_in[16];
    const float* wk   = (const float*)d_in[17];
    const float* wv   = (const float*)d_in[18];
    const float* pw   = (const float*)d_in[19];
    const float* pb   = (const float*)d_in[20];

    const size_t NQ = 8ull * 4096 * 384;
    const size_t NK = 8ull * 512 * 384;
    const size_t NW = 384ull * 384;

    char* ws = (char*)d_ws;
    size_t off = 0;
    auto carve = [&](size_t bytes) {
        void* p = ws + off;
        off += (bytes + 255) & ~(size_t)255;
        return p;
    };
    bf16* qpre = (bf16*)carve(NQ * 2);
    bf16* kpre = (bf16*)carve(NK * 2);
    bf16* vpre = (bf16*)carve(NK * 2);
    bf16* wq_b = (bf16*)carve(NW * 2);
    bf16* wk_b = (bf16*)carve(NW * 2);
    bf16* wv_b = (bf16*)carve(NW * 2);
    bf16* wp_b = (bf16*)carve(NW * 2);
    bf16* qh   = (bf16*)carve(NQ * 2);
    bf16* kh   = (bf16*)carve(NK * 2);
    bf16* vt   = (bf16*)carve(NK * 2);
    bf16* obuf = (bf16*)carve(NQ * 2);

    // 1) depthwise conv + BN -> bf16
    dwconv_bn_kernel<<<(unsigned)((NQ + 255) / 256), 256, 0, stream>>>(
        x, cqw, bq_g, bq_b, bq_m, bq_v, qpre, 1, 16);
    dwconv_bn_kernel<<<(unsigned)((NK + 255) / 256), 256, 0, stream>>>(
        x, ckw, bk_g, bk_b, bk_m, bk_v, kpre, 2, 8);
    dwconv_bn_kernel<<<(unsigned)((NK + 255) / 256), 256, 0, stream>>>(
        x, cvw, bv_g, bv_b, bv_m, bv_v, vpre, 2, 8);

    // 2) weights -> bf16
    f32_to_bf16_kernel<<<(unsigned)((NW + 255) / 256), 256, 0, stream>>>(wq, wq_b, (int)NW);
    f32_to_bf16_kernel<<<(unsigned)((NW + 255) / 256), 256, 0, stream>>>(wk, wk_b, (int)NW);
    f32_to_bf16_kernel<<<(unsigned)((NW + 255) / 256), 256, 0, stream>>>(wv, wv_b, (int)NW);
    f32_to_bf16_kernel<<<(unsigned)((NW + 255) / 256), 256, 0, stream>>>(pw, wp_b, (int)NW);

    // 3) projections (WMMA GEMMs, LDS-staged weights)
    dim3 blk(256);
    gemm_nt_wmma<0><<<dim3(6, 32768 / 128), blk, 0, stream>>>(
        qpre, wq_b, qh, nullptr, 32768, 384, 384);
    gemm_nt_wmma<0><<<dim3(6, 4096 / 128), blk, 0, stream>>>(
        kpre, wk_b, kh, nullptr, 4096, 384, 384);
    gemm_nt_wmma<1><<<dim3(6, 4096 / 128), blk, 0, stream>>>(
        vpre, wv_b, vt, nullptr, 4096, 384, 384);

    // 4) flash attention: 8*6*64 blocks of 4 waves; 64 KB dynamic LDS
    attn_kernel<<<8 * 6 * 64, 128, 2 * (128 * 64 + 64 * 128) * 2, stream>>>(
        qh, kh, vt, obuf);

    // 5) output projection + bias -> f32 d_out
    gemm_nt_wmma<2><<<dim3(6, 32768 / 128), blk, 0, stream>>>(
        obuf, wp_b, d_out, pb, 32768, 384, 384);
}
